// PINN_Conv_CNP_45930380263688
// MI455X (gfx1250) — compile-verified, matrix operand order checked by reading, not run on hardware
//
#include <hip/hip_runtime.h>
#include <hip/hip_bf16.h>
#include <math.h>

typedef __attribute__((ext_vector_type(16))) _Float16 v16h;
typedef __attribute__((ext_vector_type(8)))  _Float16 v8h;
typedef __attribute__((ext_vector_type(8)))  float    v8f;
typedef __attribute__((ext_vector_type(8)))  int      vi8;

static constexpr int BATCH = 2;
static constexpr int NCTX  = 20000;
static constexpr int QN    = 4096;
static constexpr int LATC  = 16;
static constexpr int NFRQ  = 64;
static constexpr int PADW  = 68;          // 64 + 2*2 halo
static constexpr int PADSZ = PADW * PADW; // 4624 elements per channel plane
static constexpr int KBLK  = 320;         // K-block staged in LDS per round (10 chunks)

static __device__ __forceinline__ float gcoord(int i) { return -1.f + 2.f * (float)i / 63.f; }

// ---------------------------------------------------------------------------
// Kernel 0: zero the padded f16 activation buffer (halos stay 0 forever after)
// ---------------------------------------------------------------------------
__global__ void zero_f16_kernel(_Float16* __restrict__ p, int n_dwords) {
  int i = blockIdx.x * 256 + threadIdx.x;
  if (i < n_dwords) ((unsigned int*)p)[i] = 0u;
}

// ---------------------------------------------------------------------------
// Kernel 1: Gaussian scatter of context points onto padded f16 grid (B,3,68,68)
// ---------------------------------------------------------------------------
__global__ void grid_build_kernel(const float* __restrict__ xs, const float* __restrict__ ys,
                                  const float* __restrict__ vals, const float* __restrict__ ts,
                                  const unsigned char* __restrict__ mask,
                                  _Float16* __restrict__ Xpad) {
  __shared__ float sx[256], sy[256], sv[256], st[256], sm[256];
  int tid  = threadIdx.x;
  int cell = blockIdx.x * 256 + tid;           // 0..8191
  int b    = cell >> 12;
  int g    = cell & 4095;
  int x    = g & 63, y = g >> 6;
  float gx = gcoord(x);
  float gy = gcoord(y);
  float den = 0.f, wv = 0.f, wt = 0.f;
  int nch = (NCTX + 255) / 256;
  for (int ch = 0; ch < nch; ++ch) {
    int idx = ch * 256 + tid;
    bool ok = idx < NCTX;
    int src = b * NCTX + (ok ? idx : 0);
    sx[tid] = 2.f * (xs[src] * 0.01f) - 1.f;
    sy[tid] = 2.f * (ys[src] * 0.01f) - 1.f;
    sv[tid] = vals[src];
    st[tid] = ts[src];
    sm[tid] = (ok && mask[src] == 0) ? 1.f : 0.f;
    __syncthreads();
    #pragma unroll 4
    for (int j = 0; j < 256; ++j) {
      float dx = sx[j] - gx, dy = sy[j] - gy;
      float w = sm[j] * __expf(-50.f * (dx * dx + dy * dy));   // 0.5/sigma^2 = 50
      den += w; wv += w * sv[j]; wt += w * st[j];
    }
    __syncthreads();
  }
  float inv = 1.f / (den + 1e-5f);
  int pos = (y + 2) * PADW + (x + 2);
  Xpad[(b * 3 + 0) * PADSZ + pos] = (_Float16)den;
  Xpad[(b * 3 + 1) * PADSZ + pos] = (_Float16)(wv * inv);
  Xpad[(b * 3 + 2) * PADSZ + pos] = (_Float16)(wt * inv);
}

// ---------------------------------------------------------------------------
// Kernel 2: 5x5 conv as implicit GEMM with WMMA f16 -> f32.
// Each wave: 16 pixels x (NT*16) couts, NT accumulators -> one im2col gather
// feeds NT WMMAs. Weight panel (all couts, KBLK k-values) staged fragment-major
// in LDS per K-block; koff[] gives zero-check-free tap offsets into padded f16.
// ---------------------------------------------------------------------------
template <int NT>
__global__ void conv_wmma_kernel(const _Float16* __restrict__ Xpad, const float* __restrict__ W,
                                 const float* __restrict__ bias, float* __restrict__ Y,
                                 int Cin, int Cout, int K, int Kpad) {
  __shared__ __align__(32) _Float16 Wf[KBLK * 64];   // fragment-major panel (40 KB max)
  __shared__ __align__(32) int      koff[1600];      // im2col tap offsets (6.25 KB)
  int tid = threadIdx.x;
  // im2col offset table: k -> cin*PADSZ + ky*PADW + kx (k>=K points at 0; weight is 0)
  for (int k = tid; k < Kpad; k += 256) {
    int off = 0;
    if (k < K) {
      int cin = k / 25, t = k - cin * 25;
      int ky = t / 5, kx = t - ky * 5;
      off = cin * PADSZ + ky * PADW + kx;
    }
    koff[k] = off;
  }

  int wave = tid >> 5, lane = tid & 31;
  int row0 = (blockIdx.x * 8 + wave) * 16;     // 512 pixel-tiles total
  int mA = lane & 15;
  int kbase = (lane >> 4) * 8;
  int p = row0 + mA;
  int b = p >> 12, pix = p & 4095;
  int y = pix >> 6, x = pix & 63;
  int pbase = b * Cin * PADSZ + y * PADW + x;  // tap addr = pbase + koff[k]

  v8f acc[NT];
  #pragma unroll
  for (int t = 0; t < NT; ++t) acc[t] = (v8f){};

  for (int kb = 0; kb < Kpad; kb += KBLK) {
    int kbn = min(KBLK, Kpad - kb);
    __syncthreads();                            // safe to restage panel
    // stage weights fragment-major: [kc32_local][tile][lane][j]
    int total = kbn * Cout;
    for (int idx = tid; idx < total; idx += 256) {
      int j     = idx & 15;
      int lane2 = (idx >> 4) & 31;
      int rest  = idx >> 9;                     // = kc32l*NT + t
      int t     = rest % NT;
      int kc32l = rest / NT;
      int k = kb + kc32l * 32 + ((j < 8) ? 0 : 16) + (lane2 >> 4) * 8 + (j & 7);
      int n = t * 16 + (lane2 & 15);
      Wf[idx] = (_Float16)((k < K) ? W[n * K + k] : 0.f);
    }
    __syncthreads();

    int nkc = kbn >> 5;
    for (int kc32l = 0; kc32l < nkc; ++kc32l) {
      int kc = kb + kc32l * 32;
      vi8 klo = *(const vi8*)&koff[kc + kbase];
      vi8 khi = *(const vi8*)&koff[kc + 16 + kbase];
      v16h a;
      #pragma unroll
      for (int j = 0; j < 8; ++j) a[j]     = Xpad[pbase + klo[j]];
      #pragma unroll
      for (int j = 0; j < 8; ++j) a[8 + j] = Xpad[pbase + khi[j]];
      #pragma unroll
      for (int t = 0; t < NT; ++t) {
        v16h bm = *(const v16h*)&Wf[((kc32l * NT + t) * 32 + lane) * 16];
        acc[t] = __builtin_amdgcn_wmma_f32_16x16x32_f16(false, a, false, bm,
                                                        (short)0, acc[t], false, false);
      }
    }
  }

  int ncol = lane & 15;
  int mofs = (lane >> 4) * 8;
  #pragma unroll
  for (int t = 0; t < NT; ++t) {
    float bi = bias[t * 16 + ncol];
    #pragma unroll
    for (int r = 0; r < 8; ++r) {
      int pp = row0 + r + mofs;
      int bb = pp >> 12, pixp = pp & 4095;
      Y[((bb * Cout + t * 16 + ncol) << 12) + pixp] = acc[t][r] + bi;
    }
  }
}

// ---------------------------------------------------------------------------
// Kernel 3: per-channel BN stats -> folded scale/shift
// ---------------------------------------------------------------------------
__global__ void bn_stats_kernel(const float* __restrict__ Y, const float* __restrict__ g,
                                const float* __restrict__ bb, float* __restrict__ ss, int Cout) {
  __shared__ float s1[256], s2[256];
  int c = blockIdx.x, tid = threadIdx.x;
  float a = 0.f, aq = 0.f;
  for (int b = 0; b < BATCH; ++b) {
    const float* p = Y + ((b * Cout + c) << 12);
    for (int i = tid; i < 4096; i += 256) { float v = p[i]; a += v; aq += v * v; }
  }
  s1[tid] = a; s2[tid] = aq; __syncthreads();
  for (int s = 128; s > 0; s >>= 1) {
    if (tid < s) { s1[tid] += s1[tid + s]; s2[tid] += s2[tid + s]; }
    __syncthreads();
  }
  if (tid == 0) {
    float inv_n = 1.f / (float)(BATCH * 4096);
    float m = s1[0] * inv_n;
    float v = s2[0] * inv_n - m * m;
    float sc = g[c] * rsqrtf(v + 1e-5f);
    ss[c * 2 + 0] = sc;
    ss[c * 2 + 1] = bb[c] - m * sc;
  }
}

// ---------------------------------------------------------------------------
// Kernel 4: BN apply + activation -> padded f16 buffer interior
// ---------------------------------------------------------------------------
__global__ void bn_act_kernel(const float* __restrict__ Y, const float* __restrict__ ss,
                              _Float16* __restrict__ Xpad, int Cout, int use_tanh) {
  int i = blockIdx.x * 256 + threadIdx.x;      // over B*Cout*4096
  int c = (i >> 12) % Cout;
  int bc = i >> 12;                            // b*Cout + c
  int pix = i & 4095;
  float v = Y[i] * ss[c * 2] + ss[c * 2 + 1];
  v = use_tanh ? tanhf(v) : fmaxf(v, 0.f);
  Xpad[bc * PADSZ + ((pix >> 6) + 2) * PADW + (pix & 63) + 2] = (_Float16)v;
}

// ---------------------------------------------------------------------------
// Kernel 5: bilinear sample of padded f16 latent + Fourier embedding -> h0 f16
// cols: [lat 0..15 | sin 16..79 | cos 80..143 | qx,qy | zero pad to 160]
// ---------------------------------------------------------------------------
__global__ void embed_sample_kernel(const _Float16* __restrict__ latent, const float* __restrict__ qxs,
                                    const float* __restrict__ qys, const float* __restrict__ qts,
                                    const float* __restrict__ fB, _Float16* __restrict__ h0) {
  int i = blockIdx.x * 256 + threadIdx.x;      // 0..8191
  int b = i >> 12, q = i & 4095;
  float qx = qxs[b * QN + q], qy = qys[b * QN + q], qt = qts[b * QN + q];
  float nx = 2.f * (qx * 0.01f) - 1.f, ny = 2.f * (qy * 0.01f) - 1.f;
  float ix = (nx + 1.f) * 0.5f * 63.f, iy = (ny + 1.f) * 0.5f * 63.f;
  float ix0f = floorf(ix), iy0f = floorf(iy);
  float fx = ix - ix0f, fy = iy - iy0f;
  int ix0 = min(max((int)ix0f, 0), 63), ix1 = min(max((int)ix0f + 1, 0), 63);
  int iy0 = min(max((int)iy0f, 0), 63), iy1 = min(max((int)iy0f + 1, 0), 63);
  _Float16* row = h0 + (size_t)i * 160;
  #pragma unroll
  for (int c = 0; c < LATC; ++c) {
    const _Float16* img = latent + (b * LATC + c) * PADSZ;
    float v00 = (float)img[(iy0 + 2) * PADW + ix0 + 2];
    float v01 = (float)img[(iy0 + 2) * PADW + ix1 + 2];
    float v10 = (float)img[(iy1 + 2) * PADW + ix0 + 2];
    float v11 = (float)img[(iy1 + 2) * PADW + ix1 + 2];
    row[c] = (_Float16)(v00 * (1 - fx) * (1 - fy) + v01 * fx * (1 - fy) +
                        v10 * (1 - fx) * fy + v11 * fx * fy);
  }
  float t = qt * 0.1f;
  for (int f = 0; f < NFRQ; ++f) {
    float pr = 6.283185307179586f * t * fB[f];
    row[16 + f] = (_Float16)__sinf(pr);
    row[80 + f] = (_Float16)__cosf(pr);
  }
  row[144] = (_Float16)qx; row[145] = (_Float16)qy;
  for (int k = 146; k < 160; ++k) row[k] = (_Float16)0.f;
}

// ---------------------------------------------------------------------------
// Kernel 6: decoder GEMM with WMMA: Y = tanh(X @ W + b), all-f16 activations.
// Wave tile 32 rows x 16 cols: two WMMAs share one B-fragment LDS read.
// A-frags are contiguous 128-bit global loads; next chunk prefetched.
// ---------------------------------------------------------------------------
__global__ void dec_gemm_kernel(const _Float16* __restrict__ X, const float* __restrict__ W,
                                const float* __restrict__ bias, _Float16* __restrict__ Y,
                                int K, int Kpad, int N) {
  __shared__ __align__(32) _Float16 Wf[160 * 128];   // 40 KB max
  int tid = threadIdx.x;
  int NW = N >> 4;
  for (int idx = tid; idx < Kpad * N; idx += blockDim.x) {
    int j    = idx & 15;
    int lane = (idx >> 4) & 31;
    int rest = idx >> 9;                 // = kc32*NW + w
    int w    = rest % NW;
    int kc32 = rest / NW;
    int k = kc32 * 32 + ((j < 8) ? 0 : 16) + (lane >> 4) * 8 + (j & 7);
    int n = w * 16 + (lane & 15);
    Wf[idx] = (_Float16)((k < K) ? W[k * N + n] : 0.f);
  }
  __syncthreads();

  int wave = tid >> 5, lane = tid & 31;
  int n0 = wave * 16;
  int row0 = blockIdx.x * 32;                  // 32 rows per block
  int mA = lane & 15, kbase = (lane >> 4) * 8;
  const _Float16* xrow0 = X + (size_t)(row0 + mA) * Kpad;
  const _Float16* xrow1 = X + (size_t)(row0 + 16 + mA) * Kpad;

  v8f acc0 = {}, acc1 = {};
  int nk = Kpad >> 5;
  for (int kc32 = 0; kc32 < nk; ++kc32) {
    int kc = kc32 * 32;
    __builtin_prefetch(xrow0 + kc + 32, 0, 1); // pull next chunk toward WGP
    __builtin_prefetch(xrow1 + kc + 32, 0, 1);
    v8h lo0 = *(const v8h*)(xrow0 + kc + kbase);
    v8h hi0 = *(const v8h*)(xrow0 + kc + 16 + kbase);
    v8h lo1 = *(const v8h*)(xrow1 + kc + kbase);
    v8h hi1 = *(const v8h*)(xrow1 + kc + 16 + kbase);
    v16h a0 = __builtin_shufflevector(lo0, hi0, 0, 1, 2, 3, 4, 5, 6, 7,
                                                8, 9, 10, 11, 12, 13, 14, 15);
    v16h a1 = __builtin_shufflevector(lo1, hi1, 0, 1, 2, 3, 4, 5, 6, 7,
                                                8, 9, 10, 11, 12, 13, 14, 15);
    v16h bm = *(const v16h*)&Wf[((kc32 * NW + wave) * 32 + lane) * 16];
    acc0 = __builtin_amdgcn_wmma_f32_16x16x32_f16(false, a0, false, bm, (short)0, acc0, false, false);
    acc1 = __builtin_amdgcn_wmma_f32_16x16x32_f16(false, a1, false, bm, (short)0, acc1, false, false);
  }

  int ncol = lane & 15, mofs = (lane >> 4) * 8;
  float bi = bias[n0 + ncol];
  #pragma unroll
  for (int r = 0; r < 8; ++r) {
    Y[(size_t)(row0 + r + mofs) * N + n0 + ncol]      = (_Float16)tanhf(acc0[r] + bi);
    Y[(size_t)(row0 + 16 + r + mofs) * N + n0 + ncol] = (_Float16)tanhf(acc1[r] + bi);
  }
}

// ---------------------------------------------------------------------------
// Kernel 7: four tiny heads from h (8192x64 f16) -> out (B,Q,6) f32
// ---------------------------------------------------------------------------
__global__ void heads_kernel(const _Float16* __restrict__ H,
                             const float* __restrict__ huw, const float* __restrict__ hub,
                             const float* __restrict__ hfw, const float* __restrict__ hfb,
                             const float* __restrict__ hmw, const float* __restrict__ hmb,
                             const float* __restrict__ hsw, const float* __restrict__ hsb,
                             float* __restrict__ out) {
  int i = blockIdx.x * 256 + threadIdx.x;      // 0..8191
  const _Float16* h = H + (size_t)i * 64;
  float u0 = 0, u1 = 0, f0 = 0, f1 = 0, mu = 0, sd = 0;
  #pragma unroll 4
  for (int k = 0; k < 64; ++k) {
    float v = (float)h[k];
    u0 += v * huw[k * 2];  u1 += v * huw[k * 2 + 1];
    f0 += v * hfw[k * 2];  f1 += v * hfw[k * 2 + 1];
    mu += v * hmw[k];      sd += v * hsw[k];
  }
  u0 += hub[0]; u1 += hub[1]; f0 += hfb[0]; f1 += hfb[1]; mu += hmb[0]; sd += hsb[0];
  float sp = (sd > 20.f) ? sd : log1pf(__expf(sd));
  out[i * 6 + 0] = mu;
  out[i * 6 + 1] = sp + 1e-6f;
  out[i * 6 + 2] = u0; out[i * 6 + 3] = u1;
  out[i * 6 + 4] = f0; out[i * 6 + 5] = f1;
}

// ---------------------------------------------------------------------------
extern "C" void kernel_launch(void* const* d_in, const int* in_sizes, int n_in,
                              void* d_out, int out_size, void* d_ws, size_t ws_size,
                              hipStream_t stream) {
  const float* ctx_xs = (const float*)d_in[0];
  const float* ctx_ys = (const float*)d_in[1];
  const float* ctx_v  = (const float*)d_in[2];
  const float* ctx_ts = (const float*)d_in[3];
  const unsigned char* ctx_mask = (const unsigned char*)d_in[4];
  const float* q_xs = (const float*)d_in[5];
  const float* q_ys = (const float*)d_in[6];
  const float* q_ts = (const float*)d_in[7];
  const float* fourierB = (const float*)d_in[44];

  // workspace carve-up (byte offsets, all 256B-aligned)
  char* base = (char*)d_ws;
  float*    bufB  = (float*)(base);                      // conv raw f32: 2,097,152 B
  float*    stats = (float*)(base + 2097152);            // 1 KB (pad to 2,098,176)
  _Float16* Xpad  = (_Float16*)(base + 2098176);         // B*64*4624 f16: 1,183,744 B
  _Float16* h0    = (_Float16*)(base + 3281920);         // 8192*160 f16: 2,621,440 B
  _Float16* hA    = (_Float16*)(base + 5903360);         // 8192*128 f16: 2,097,152 B
  _Float16* hB    = (_Float16*)(base + 8000512);         // 8192*128 f16: 2,097,152 B

  // zero padded activation buffer once; halos stay zero for every layer
  int n_dwords = BATCH * 64 * PADSZ / 2;                 // 295,936
  zero_f16_kernel<<<(n_dwords + 255) / 256, 256, 0, stream>>>(Xpad, n_dwords);

  grid_build_kernel<<<32, 256, 0, stream>>>(ctx_xs, ctx_ys, ctx_v, ctx_ts, ctx_mask, Xpad);

  const int cins[5]  = {3, 64, 64, 64, 64};
  const int couts[5] = {64, 64, 64, 64, 16};
  for (int L = 0; L < 5; ++L) {
    const float* w  = (const float*)d_in[8 + 4 * L];
    const float* cb = (const float*)d_in[9 + 4 * L];
    const float* bg = (const float*)d_in[10 + 4 * L];
    const float* bb = (const float*)d_in[11 + 4 * L];
    int K = cins[L] * 25;
    int Kpad = ((K + 31) / 32) * 32;
    if (couts[L] == 64)
      conv_wmma_kernel<4><<<64, 256, 0, stream>>>(Xpad, w, cb, bufB, cins[L], couts[L], K, Kpad);
    else
      conv_wmma_kernel<1><<<64, 256, 0, stream>>>(Xpad, w, cb, bufB, cins[L], couts[L], K, Kpad);
    bn_stats_kernel<<<couts[L], 256, 0, stream>>>(bufB, bg, bb, stats, couts[L]);
    int nelem = BATCH * couts[L] * 4096;
    bn_act_kernel<<<nelem / 256, 256, 0, stream>>>(bufB, stats, Xpad, couts[L], (L == 4) ? 1 : 0);
  }

  // latent = Xpad (B,16,68,68) f16
  embed_sample_kernel<<<32, 256, 0, stream>>>(Xpad, q_xs, q_ys, q_ts, fourierB, h0);

  dec_gemm_kernel<<<256, 256, 0, stream>>>(h0, (const float*)d_in[28], (const float*)d_in[29], hA, 146, 160, 128);
  dec_gemm_kernel<<<256, 256, 0, stream>>>(hA, (const float*)d_in[30], (const float*)d_in[31], hB, 128, 128, 128);
  dec_gemm_kernel<<<256, 256, 0, stream>>>(hB, (const float*)d_in[32], (const float*)d_in[33], hA, 128, 128, 128);
  dec_gemm_kernel<<<256, 128, 0, stream>>>(hA, (const float*)d_in[34], (const float*)d_in[35], hB, 128, 128, 64);

  heads_kernel<<<32, 256, 0, stream>>>(hB,
      (const float*)d_in[36], (const float*)d_in[37],
      (const float*)d_in[38], (const float*)d_in[39],
      (const float*)d_in[40], (const float*)d_in[41],
      (const float*)d_in[42], (const float*)d_in[43],
      (float*)d_out);
}
